// FirstModule_34179349742204
// MI455X (gfx1250) — matrix operand order repaired and checked by reading, not compiled, not run on hardware
//
#include <hip/hip_runtime.h>
#include <cstddef>

// ---------------------------------------------------------------- types
typedef __attribute__((ext_vector_type(16))) __bf16        v16bf;
typedef __attribute__((ext_vector_type(8)))  float         v8f;
typedef __attribute__((ext_vector_type(4)))  unsigned int  u32x4;
typedef __attribute__((ext_vector_type(4)))  unsigned int  uv4;
typedef __attribute__((ext_vector_type(8)))  int           iv8;
typedef __attribute__((ext_vector_type(4)))  int           iv4;
typedef unsigned short u16;

#define SEQ    4096
#define DM     1024          // D
#define HID    1024          // H
#define KA     2048          // 2*D  (K of GEMM1)
#define N1     3072          // 3*H  (N of GEMM1)
#define CHUNK  128
#define NCHUNK 32            // SEQ / CHUNK

#if defined(__gfx1250__) && __has_builtin(__builtin_amdgcn_tensor_load_to_lds) && \
    __has_builtin(__builtin_amdgcn_s_wait_tensorcnt)
#define HAS_TDM 1
#else
#define HAS_TDM 0
#endif

// ---------------------------------------------------------------- helpers
static __device__ __forceinline__ u16 f2bf(float f) {
  union { float f; unsigned int u; } v; v.f = f;
  unsigned int r = v.u + 0x7FFFu + ((v.u >> 16) & 1u);   // round-to-nearest-even
  return (u16)(r >> 16);
}
static __device__ __forceinline__ float bf2f(u16 h) {
  union { unsigned int u; float f; } v; v.u = ((unsigned int)h) << 16;
  return v.f;
}
static __device__ __forceinline__ void split_bf(float f, u16& hi, u16& lo) {
  hi = f2bf(f);
  lo = f2bf(f - bf2f(hi));
}
static __device__ __forceinline__ float sigmoid_fast(float x) {
  return 1.0f / (1.0f + __expf(-x));
}

#if HAS_TDM
// LDS byte offset of a pointer into __shared__ memory (addrspace(3) view).
static __device__ __forceinline__ unsigned lds_off(const void* p) {
  return (unsigned)(unsigned long long)(__attribute__((address_space(3))) const void*)p;
}
static __device__ __forceinline__ void tdm_issue(uv4 g0, iv8 g1) {
  iv4 z4 = (iv4)0;
#if __clang_major__ >= 23
  iv8 z8 = (iv8)0;
  __builtin_amdgcn_tensor_load_to_lds(g0, g1, z4, z4, z8, 0);
#else
  __builtin_amdgcn_tensor_load_to_lds(g0, g1, z4, z4, 0);
#endif
}
// Issue a TDM load of a 2-D tile: tile_rows x 32 bf16 elements out of a
// [nrows_tensor x K] row-major u16 tensor starting at g, into LDS at lds.
// D# layout per CDNA5 ISA 8.3/8.4: count=1, type=2 (image), data_size=1 (2B),
// tensor_dim0=K, tensor_dim0_stride=K, tile_dim0=32, tile_dim1=tile_rows.
static __device__ __forceinline__ void tdm_tile_2d(const u16* g, unsigned lds,
                                                   int K, int nrows_tensor,
                                                   int tile_rows) {
  const unsigned long long ga = (unsigned long long)g;
  uv4 g0;
  g0[0] = 1u;                                              // count=1 (valid D#)
  g0[1] = lds;                                             // lds_addr (bytes)
  g0[2] = (unsigned)ga;                                    // global_addr[31:0]
  g0[3] = ((unsigned)(ga >> 32) & 0x01FFFFFFu) | 0x80000000u; // addr[56:32] | type=2
  iv8 g1;
  g1[0] = 1 << 16;                                         // data_size = 2 bytes
  g1[1] = (int)(((unsigned)K & 0xFFFFu) << 16);            // tensor_dim0[15:0]
  g1[2] = (int)((((unsigned)K >> 16) & 0xFFFFu) |
                (((unsigned)nrows_tensor & 0xFFFFu) << 16)); // dim0 hi | dim1 lo
  g1[3] = (int)((((unsigned)nrows_tensor >> 16) & 0xFFFFu) |
                (32u << 16));                              // dim1 hi | tile_dim0=32
  g1[4] = tile_rows;                                       // tile_dim1 (tile_dim2=0)
  g1[5] = K;                                               // tensor_dim0_stride lo32
  g1[6] = 0;                                               // stride0 hi | stride1 lo
  g1[7] = 0;                                               // stride1 hi
  tdm_issue(g0, g1);
}
#endif

// ---------------------------------------------------------------- prefix-sum stage
__global__ __launch_bounds__(256) void k_chunksum(const float* __restrict__ x,
                                                  float* __restrict__ csum) {
  const int d = blockIdx.x * 256 + threadIdx.x;   // column
  const int c = blockIdx.y;                       // chunk
  const float* p = x + (size_t)c * CHUNK * DM + d;
  float a = 0.f;
  #pragma unroll 8
  for (int r = 0; r < CHUNK; ++r) a += p[(size_t)r * DM];
  csum[c * DM + d] = a;
}

__global__ __launch_bounds__(256) void k_chunkscan(float* __restrict__ csum) {
  const int d = blockIdx.x * 256 + threadIdx.x;
  float acc = 0.f;
  #pragma unroll
  for (int c = 0; c < NCHUNK; ++c) {
    const int i = c * DM + d;
    const float t = csum[i];
    csum[i] = acc;
    acc += t;
  }
}

__global__ __launch_bounds__(256) void k_emit(const float* __restrict__ x,
                                              const float* __restrict__ csum,
                                              u16* __restrict__ Ahi,
                                              u16* __restrict__ Alo) {
  const int d = blockIdx.x * 256 + threadIdx.x;
  const int c = blockIdx.y;
  float acc = csum[c * DM + d];                    // exclusive offset entering chunk
  #pragma unroll 4
  for (int r = 0; r < CHUNK; ++r) {
    const int t = c * CHUNK + r;
    const float xv = x[(size_t)t * DM + d];
    u16 hs, ls, hx, lx;
    split_bf(acc, hs, ls);                         // s[t] (exclusive)
    split_bf(xv,  hx, lx);
    const size_t rowb = (size_t)t * KA;
    Ahi[rowb + d]      = hx;  Alo[rowb + d]      = lx;
    Ahi[rowb + DM + d] = hs;  Alo[rowb + DM + d] = ls;
    acc += xv;
  }
}

// ---------------------------------------------------------------- weight split
__global__ __launch_bounds__(256) void k_split(const float* __restrict__ src,
                                               u16* __restrict__ hi,
                                               u16* __restrict__ lo) {
  const size_t i = (size_t)blockIdx.x * 256 + threadIdx.x;
  u16 h, l;
  split_bf(src[i], h, l);
  hi[i] = h; lo[i] = l;
}

// ---------------------------------------------------------------- WMMA GEMM
// C[M,N] = A[M,K] * B[N,K]^T with split-bf16 operands (hi*hi + hi*lo + lo*hi).
// Block tile BM=128, BN=64, BK=32; 8 wave32 waves (4Mx2N), 32x32 per wave.
// TDM path: double-buffered LDS, wave 0 DMAs the next K-tile (4 tensor_load_to_lds)
// while all waves compute; TENSORcnt + barrier close the pipeline.
union Frag { u32x4 u[2]; v16bf v; };

template <int MODE>
__global__ __launch_bounds__(256) void k_wmma_gemm(
    const u16* __restrict__ Ahi, const u16* __restrict__ Alo,
    const u16* __restrict__ Bhi, const u16* __restrict__ Blo,
    const float* __restrict__ bias,
    float* __restrict__ out_fg, u16* __restrict__ thhi, u16* __restrict__ thlo,
    float* __restrict__ ig, float* __restrict__ out_hr, int K) {
  __shared__ __align__(16) u16 sAhi[2][128 * 32];
  __shared__ __align__(16) u16 sAlo[2][128 * 32];
  __shared__ __align__(16) u16 sBhi[2][64 * 32];
  __shared__ __align__(16) u16 sBlo[2][64 * 32];

  const int tid  = threadIdx.x;
  const int lane = tid & 31;
  const int wave = tid >> 5;              // wave32: 8 waves / block
  const int wm   = wave >> 1;             // 0..3  (M direction)
  const int wn   = wave & 1;              // 0..1  (N direction)
  const int hsel = lane >> 4;             // half-wave select
  const int l16  = lane & 15;
  const int blockN = blockIdx.x * 64;
  const int blockM = blockIdx.y * 128;

  const v8f vz = {0.f, 0.f, 0.f, 0.f, 0.f, 0.f, 0.f, 0.f};
  v8f acc[2][2];
  acc[0][0] = vz; acc[0][1] = vz; acc[1][0] = vz; acc[1][1] = vz;

#if HAS_TDM
  const int nrowsB = (int)gridDim.x * 64;
  if (tid < 32) {                         // prologue: DMA K-tile 0 into buffer 0
    tdm_tile_2d(&Ahi[(size_t)blockM * K], lds_off(&sAhi[0][0]), K, SEQ, 128);
    tdm_tile_2d(&Alo[(size_t)blockM * K], lds_off(&sAlo[0][0]), K, SEQ, 128);
    tdm_tile_2d(&Bhi[(size_t)blockN * K], lds_off(&sBhi[0][0]), K, nrowsB, 64);
    tdm_tile_2d(&Blo[(size_t)blockN * K], lds_off(&sBlo[0][0]), K, nrowsB, 64);
  }
#endif

  for (int k0 = 0; k0 < K; k0 += 32) {
    const int cur = (k0 >> 5) & 1;

#if HAS_TDM
    if (tid < 32) __builtin_amdgcn_s_wait_tensorcnt(0);   // buffer `cur` landed
    __syncthreads();                                      // (also waits DScnt)
    if (k0 + 32 < K && tid < 32) {                        // DMA next tile -> nxt
      const int nxt = cur ^ 1;
      const size_t gA = (size_t)blockM * K + k0 + 32;
      const size_t gB = (size_t)blockN * K + k0 + 32;
      tdm_tile_2d(&Ahi[gA], lds_off(&sAhi[nxt][0]), K, SEQ, 128);
      tdm_tile_2d(&Alo[gA], lds_off(&sAlo[nxt][0]), K, SEQ, 128);
      tdm_tile_2d(&Bhi[gB], lds_off(&sBhi[nxt][0]), K, nrowsB, 64);
      tdm_tile_2d(&Blo[gB], lds_off(&sBlo[nxt][0]), K, nrowsB, 64);
    }
#else
    __syncthreads();                                      // sync fallback path
    #pragma unroll
    for (int it = 0; it < 2; ++it) {                      // A: 512 vectors / plane
      const int v   = it * 256 + tid;
      const int row = v >> 2;
      const int kc  = v & 3;
      const size_t g = (size_t)(blockM + row) * K + k0 + kc * 8;
      *(u32x4*)&sAhi[cur][row * 32 + kc * 8] = *(const u32x4*)&Ahi[g];
      *(u32x4*)&sAlo[cur][row * 32 + kc * 8] = *(const u32x4*)&Alo[g];
      if (k0 + 32 < K) {
        __builtin_prefetch(&Ahi[g + 32], 0, 0);
        __builtin_prefetch(&Alo[g + 32], 0, 0);
      }
    }
    {                                                     // B: 256 vectors / plane
      const int row = tid >> 2;
      const int kc  = tid & 3;
      const size_t g = (size_t)(blockN + row) * K + k0 + kc * 8;
      *(u32x4*)&sBhi[cur][row * 32 + kc * 8] = *(const u32x4*)&Bhi[g];
      *(u32x4*)&sBlo[cur][row * 32 + kc * 8] = *(const u32x4*)&Blo[g];
      if (k0 + 32 < K) {
        __builtin_prefetch(&Bhi[g + 32], 0, 0);
        __builtin_prefetch(&Blo[g + 32], 0, 0);
      }
    }
    __syncthreads();
#endif

    // ---- load fragments from buffer `cur` (layouts per ISA 7.12.2) ----
    Frag fah[2], fal[2], fbh[2], fbl[2];
    #pragma unroll
    for (int mi = 0; mi < 2; ++mi) {
      const int row = wm * 32 + mi * 16 + l16;
      const u32x4* ph = (const u32x4*)&sAhi[cur][row * 32 + hsel * 8];
      const u32x4* pl = (const u32x4*)&sAlo[cur][row * 32 + hsel * 8];
      fah[mi].u[0] = ph[0]; fah[mi].u[1] = ph[2];    // K chunks {h, h+2}
      fal[mi].u[0] = pl[0]; fal[mi].u[1] = pl[2];
    }
    #pragma unroll
    for (int ni = 0; ni < 2; ++ni) {
      const int row = wn * 32 + ni * 16 + l16;
      const u32x4* ph = (const u32x4*)&sBhi[cur][row * 32 + hsel * 8];
      const u32x4* pl = (const u32x4*)&sBlo[cur][row * 32 + hsel * 8];
      fbh[ni].u[0] = ph[0]; fbh[ni].u[1] = ph[2];    // K chunks {h, h+2}
      fbl[ni].u[0] = pl[0]; fbl[ni].u[1] = pl[2];
    }

    // ---- 12 WMMAs: hi*hi + hi*lo + lo*hi per 16x16 tile ----
    #pragma unroll
    for (int mi = 0; mi < 2; ++mi)
      #pragma unroll
      for (int ni = 0; ni < 2; ++ni) {
        acc[mi][ni] = __builtin_amdgcn_wmma_f32_16x16x32_bf16(
            false, fah[mi].v, false, fbh[ni].v, (short)0, acc[mi][ni], false, false);
        acc[mi][ni] = __builtin_amdgcn_wmma_f32_16x16x32_bf16(
            false, fah[mi].v, false, fbl[ni].v, (short)0, acc[mi][ni], false, false);
        acc[mi][ni] = __builtin_amdgcn_wmma_f32_16x16x32_bf16(
            false, fal[mi].v, false, fbh[ni].v, (short)0, acc[mi][ni], false, false);
      }
  }

  // ---- epilogue (C/D layout: VGPR e -> M=e / e+8 by half-wave, N=lane&15) ----
  const int region = blockN >> 10;   // GEMM1: 0=ig, 1=fg, 2=tanh (BN=64 divides 1024)
  #pragma unroll
  for (int mi = 0; mi < 2; ++mi) {
    #pragma unroll
    for (int ni = 0; ni < 2; ++ni) {
      const int mrow0 = blockM + wm * 32 + mi * 16 + hsel * 8;
      const int ncol  = blockN + wn * 32 + ni * 16 + l16;
      const float bv  = bias[ncol];
      const v8f c = acc[mi][ni];
      #pragma unroll
      for (int e = 0; e < 8; ++e) {
        const int row = mrow0 + e;
        const float g = c[e] + bv;
        if (MODE == 0) {
          const size_t idx = (size_t)row * HID + (ncol & 1023);
          if (region == 0) {
            ig[idx] = sigmoid_fast(g);
          } else if (region == 1) {
            out_fg[idx] = sigmoid_fast(g);
          } else {
            const float t = tanhf(g);
            u16 h, l; split_bf(t, h, l);
            thhi[idx] = h; thlo[idx] = l;
          }
        } else {
          const size_t idx = (size_t)row * HID + ncol;
          out_hr[idx] = g * ig[idx];
        }
      }
    }
  }
}

// ---------------------------------------------------------------- launch
extern "C" void kernel_launch(void* const* d_in, const int* in_sizes, int n_in,
                              void* d_out, int out_size, void* d_ws, size_t ws_size,
                              hipStream_t stream) {
  (void)in_sizes; (void)n_in; (void)out_size; (void)ws_size;
  const float* x  = (const float*)d_in[0];   // [4096,1024]
  const float* W1 = (const float*)d_in[1];   // [3072,2048]
  const float* b1 = (const float*)d_in[2];   // [3072]
  const float* W2 = (const float*)d_in[3];   // [1024,1024]
  const float* b2 = (const float*)d_in[4];   // [1024]

  char* ws = (char*)d_ws;
  size_t off = 0;
  float* csum = (float*)(ws + off); off += (size_t)NCHUNK * DM * 4;             // 128 KB
  u16* Ahi  = (u16*)(ws + off); off += (size_t)SEQ * KA * 2;                    // 16 MB
  u16* Alo  = (u16*)(ws + off); off += (size_t)SEQ * KA * 2;                    // 16 MB
  u16* W1hi = (u16*)(ws + off); off += (size_t)N1 * KA * 2;                     // 12.6 MB
  u16* W1lo = (u16*)(ws + off); off += (size_t)N1 * KA * 2;
  u16* W2hi = (u16*)(ws + off); off += (size_t)HID * HID * 2;                   // 2 MB
  u16* W2lo = (u16*)(ws + off); off += (size_t)HID * HID * 2;
  float* igbuf = (float*)(ws + off); off += (size_t)SEQ * HID * 4;              // 16 MB
  u16* thhi = (u16*)(ws + off); off += (size_t)SEQ * HID * 2;                   // 8 MB
  u16* thlo = (u16*)(ws + off); off += (size_t)SEQ * HID * 2;

  float* out_fg = (float*)d_out;                       // first output
  float* out_hr = out_fg + (size_t)SEQ * HID;          // second output

  // prefix-sum -> A planes
  k_chunksum<<<dim3(DM / 256, NCHUNK), 256, 0, stream>>>(x, csum);
  k_chunkscan<<<DM / 256, 256, 0, stream>>>(csum);
  k_emit<<<dim3(DM / 256, NCHUNK), 256, 0, stream>>>(x, csum, Ahi, Alo);

  // weight splits
  k_split<<<(N1 * KA) / 256, 256, 0, stream>>>(W1, W1hi, W1lo);
  k_split<<<(HID * HID) / 256, 256, 0, stream>>>(W2, W2hi, W2lo);

  // GEMM1: [4096,2048] x [3072,2048]^T, fused sigmoid/sigmoid/tanh epilogue
  k_wmma_gemm<0><<<dim3(N1 / 64, SEQ / 128), 256, 0, stream>>>(
      Ahi, Alo, W1hi, W1lo, b1, out_fg, thhi, thlo, igbuf, nullptr, KA);

  // GEMM2: [4096,1024] x [1024,1024]^T, fused (+b2)*ig epilogue
  k_wmma_gemm<1><<<dim3(HID / 64, SEQ / 128), 256, 0, stream>>>(
      thhi, thlo, W2hi, W2lo, b2, nullptr, nullptr, nullptr, igbuf, out_hr, HID);
}
